// SE3UnetV2_66864050864665
// MI455X (gfx1250) — compile-verified, matrix operand order checked by reading, not run on hardware
//
#include <hip/hip_runtime.h>
#include <hip/hip_bf16.h>
#include <math.h>

#define NN   65536
#define DEG  8
#define CH   32
#define EPSV 1e-8f
#define LSCALE 0.10206207261596577f   /* 1/sqrt(3*ck) = 1/sqrt(96) */

typedef __attribute__((ext_vector_type(16))) __bf16 v16bf;
typedef __attribute__((ext_vector_type(8)))  __bf16 v8bf;
typedef __attribute__((ext_vector_type(8)))  float  v8f;

static __device__ __forceinline__ __bf16 f2bf(float x) {
  unsigned u = __builtin_bit_cast(unsigned, x);
  unsigned r = u + 0x7FFFu + ((u >> 16) & 1u);      // round-to-nearest-even
  unsigned short h = (unsigned short)(r >> 16);
  return __builtin_bit_cast(__bf16, h);
}
static __device__ __forceinline__ float bf2f(__bf16 x) {
  unsigned short h = __builtin_bit_cast(unsigned short, x);
  unsigned u = ((unsigned)h) << 16;
  return __builtin_bit_cast(float, u);
}

// ---------------------------------------------------------------------------
// One-time weight pre-swizzle: 6 f32 32x32 matrices -> bf16 B-fragment layout.
// Per matrix: [colhalf][lane][16] so each wave's B operand is one contiguous
// 32-byte per-lane load.  k0 = (vv>>2)*16 + (lane>>4)*8 + (vv&3)*2.
// ---------------------------------------------------------------------------
__global__ void swizzle_w6(const float* __restrict__ W0, const float* __restrict__ W1,
                           const float* __restrict__ W2, const float* __restrict__ W3,
                           const float* __restrict__ W4, const float* __restrict__ W5,
                           __bf16* __restrict__ out)
{
  const int m = blockIdx.x;                           // matrix 0..5
  const float* W = (m == 0) ? W0 : (m == 1) ? W1 : (m == 2) ? W2
                 : (m == 3) ? W3 : (m == 4) ? W4 : W5;
  const int lane = threadIdx.x & 31;
  const int colhalf = threadIdx.x >> 5;               // blockDim 64
  const int half = lane >> 4;
  const int col = (colhalf << 4) + (lane & 15);
  __bf16* o = out + (size_t)m * 1024 + ((size_t)colhalf * 32 + lane) * 16;
#pragma unroll
  for (int vv = 0; vv < 8; ++vv) {
    const int k0 = ((vv >> 2) << 4) + (half << 3) + ((vv & 3) << 1);
    o[2 * vv]     = f2bf(W[k0 * CH + col]);
    o[2 * vv + 1] = f2bf(W[(k0 + 1) * CH + col]);
  }
}

// ---------------------------------------------------------------------------
// GEMM phase: O_w = X(bf16) * W_w(bf16, pre-swizzled), f32 accumulate.
// One 16-row tile per block; 8 waves = 4 weights x 2 column halves; each wave
// holds B in registers and issues 3 back-to-back v_wmma (one per spatial dim).
// A fragment = two contiguous 16B bf16 loads per lane (documented A layout).
// ---------------------------------------------------------------------------
__global__ void gemm4_wmma(const __bf16* __restrict__ X,    // [3][n][32] bf16
                           const __bf16* __restrict__ Bq, const __bf16* __restrict__ Bk,
                           const __bf16* __restrict__ Bv, const __bf16* __restrict__ Bs,
                           float* __restrict__ O0, float* __restrict__ O1,
                           float* __restrict__ O2, float* __restrict__ O3,
                           int n, int nw)
{
  const int lane = threadIdx.x & 31;
  const int wave = threadIdx.x >> 5;
  const int w    = wave >> 1;
  if (w >= nw) return;                                // uniform per wave
  const int colhalf = wave & 1;
  const __bf16* Bp = (w == 0) ? Bq : (w == 1) ? Bk : (w == 2) ? Bv : Bs;
  float* Obase = (w == 0) ? O0 : (w == 1) ? O1 : (w == 2) ? O2 : O3;

  const v16bf b = *(const v16bf*)(Bp + ((size_t)colhalf * 32 + lane) * 16);

  const int half = lane >> 4;
  const int r    = (blockIdx.x << 4) + (lane & 15);
  const int coln = (colhalf << 4) + (lane & 15);

#pragma unroll
  for (int d = 0; d < 3; ++d) {
    const __bf16* Xp = X + (size_t)d * n * CH + (size_t)r * CH;
    const v8bf lo = *(const v8bf*)(Xp + (half << 3));
    const v8bf hi = *(const v8bf*)(Xp + 16 + (half << 3));
    v16bf a;
#pragma unroll
    for (int e = 0; e < 8; ++e) { a[e] = lo[e]; a[8 + e] = hi[e]; }
    v8f c = {};
    c = __builtin_amdgcn_wmma_f32_16x16x32_bf16(false, a, false, b, (short)0, c, false, false);
    float* Op = Obase + (size_t)d * n * CH;
#pragma unroll
    for (int j = 0; j < 8; ++j)
      Op[(size_t)((blockIdx.x << 4) + j + (half << 3)) * CH + coln] = c[j];
  }
}

// ---------------------------------------------------------------------------
// Edge phase: one wave32 per destination node (lane = channel).
// KE = 8<<level incoming edges, contiguous; src index shifted by level.
// ---------------------------------------------------------------------------
template<int KE, int SHIFT>
__global__ void edge_attn(const float* __restrict__ Qb, const float* __restrict__ Kb,
                          const float* __restrict__ Vb, const float* __restrict__ Sb,
                          const float* __restrict__ wr, const float* __restrict__ pos,
                          const int* __restrict__ src,
                          float* __restrict__ out, int n, int cout)
{
  const int lane = threadIdx.x & 31;
  const int j = blockIdx.x * (blockDim.x >> 5) + (threadIdx.x >> 5);
  if (j >= n) return;
  const size_t pq = (size_t)n * CH;
  const float q0 = Qb[0 * pq + (size_t)j * CH + lane];
  const float q1 = Qb[1 * pq + (size_t)j * CH + lane];
  const float q2 = Qb[2 * pq + (size_t)j * CH + lane];
  const float px = pos[3 * j], py = pos[3 * j + 1], pz = pos[3 * j + 2];
  const int base = j * KE;

  float logits[KE];
  float m = -3.4e38f;
#pragma unroll
  for (int t = 0; t < KE; ++t) {
    const int s = (int)(((unsigned)src[base + t]) >> SHIFT);
    float p = q0 * Kb[0 * pq + (size_t)s * CH + lane]
            + q1 * Kb[1 * pq + (size_t)s * CH + lane]
            + q2 * Kb[2 * pq + (size_t)s * CH + lane];
#pragma unroll
    for (int off = 16; off >= 1; off >>= 1) p += __shfl_xor(p, off, 32);
    p *= LSCALE;
    logits[t] = p;
    m = fmaxf(m, p);
  }
  float denom = 0.f;
#pragma unroll
  for (int t = 0; t < KE; ++t) { logits[t] = __expf(logits[t] - m); denom += logits[t]; }
  const float inv = 1.f / (denom + EPSV);

  const bool act = lane < cout;
  const float wc = act ? wr[lane] : 0.f;
  const size_t pv = (size_t)n * cout;
  float a0 = 0.f, a1 = 0.f, a2 = 0.f;
#pragma unroll
  for (int t = 0; t < KE; ++t) {
    const int s = (int)(((unsigned)src[base + t]) >> SHIFT);
    const float rx = pos[3 * s] - px, ry = pos[3 * s + 1] - py, rz = pos[3 * s + 2] - pz;
    const float irn = rsqrtf(rx * rx + ry * ry + rz * rz + EPSV);
    const float alpha = logits[t] * inv;
    if (act) {
      a0 += alpha * (Vb[0 * pv + (size_t)s * cout + lane] + wc * rx * irn);
      a1 += alpha * (Vb[1 * pv + (size_t)s * cout + lane] + wc * ry * irn);
      a2 += alpha * (Vb[2 * pv + (size_t)s * cout + lane] + wc * rz * irn);
    }
  }
  if (act) {
    out[0 * pv + (size_t)j * cout + lane] = a0 + Sb[0 * pv + (size_t)j * cout + lane];
    out[1 * pv + (size_t)j * cout + lane] = a1 + Sb[1 * pv + (size_t)j * cout + lane];
    out[2 * pv + (size_t)j * cout + lane] = a2 + Sb[2 * pv + (size_t)j * cout + lane];
  }
}

// cin==1 layer: Q/K/Val/S are outer products of scalar-per-dim feature with 32-vec weights.
__global__ void outer1(const float* __restrict__ f,     // [3][n]
                       const float* __restrict__ Wq, const float* __restrict__ Wk,
                       const float* __restrict__ Wv, const float* __restrict__ Ws,
                       float* __restrict__ Qb, float* __restrict__ Kb,
                       float* __restrict__ Vb, float* __restrict__ Sb, int n)
{
  const int gid = blockIdx.x * blockDim.x + threadIdx.x;
  const int c = gid & 31, i = gid >> 5;
  if (i >= n) return;
#pragma unroll
  for (int d = 0; d < 3; ++d) {
    const float x = f[(size_t)d * n + i];
    const size_t o = (size_t)d * n * CH + (size_t)i * CH + c;
    Qb[o] = x * Wq[c]; Kb[o] = x * Wk[c]; Vb[o] = x * Wv[c]; Sb[o] = x * Ws[c];
  }
}

// cout==1 matvecs for the output layer: Val = X*Wv, S = X*Ws  ([n,32]x[32,1]).
__global__ void matvec2(const __bf16* __restrict__ X, const float* __restrict__ Wv,
                        const float* __restrict__ Ws,
                        float* __restrict__ Vb, float* __restrict__ Sb, int n)
{
  const int gid = blockIdx.x * blockDim.x + threadIdx.x;   // 3*n
  if (gid >= 3 * n) return;
  const int d = gid / n, i = gid - d * n;
  const __bf16* xp = X + (size_t)d * n * CH + (size_t)i * CH;
  float sv = 0.f, ss = 0.f;
#pragma unroll
  for (int c = 0; c < CH; ++c) { const float x = bf2f(xp[c]); sv += x * Wv[c]; ss += x * Ws[c]; }
  Vb[(size_t)d * n + i] = sv;
  Sb[(size_t)d * n + i] = ss;
}

// GNormBias: o = bf16( h * relu(|h|+b)/|h| ) per (node, channel) over the 3 dims.
__global__ void normbias_bf(const float* __restrict__ h, const float* __restrict__ b,
                            __bf16* __restrict__ o, int n)
{
  const int gid = blockIdx.x * blockDim.x + threadIdx.x;   // n*32
  const int c = gid & 31, i = gid >> 5;
  if (i >= n) return;
  const size_t p = (size_t)n * CH;
  const size_t off = (size_t)i * CH + c;
  const float h0 = h[off], h1 = h[p + off], h2 = h[2 * p + off];
  const float nrm = sqrtf(h0 * h0 + h1 * h1 + h2 * h2 + EPSV);
  const float g = fmaxf(nrm + b[c], 0.f) / nrm;
  o[off] = f2bf(h0 * g); o[p + off] = f2bf(h1 * g); o[2 * p + off] = f2bf(h2 * g);
}

// mean-pool 2:1 over [3][2n] planes -> [3][n]
__global__ void pool_feat(const float* __restrict__ in, float* __restrict__ out, int nOut)
{
  const int gid = blockIdx.x * blockDim.x + threadIdx.x;   // 3*nOut
  if (gid >= 3 * nOut) return;
  const int d = gid / nOut, j = gid - d * nOut;
  out[gid] = 0.5f * (in[(size_t)d * 2 * nOut + 2 * j] + in[(size_t)d * 2 * nOut + 2 * j + 1]);
}

// mean-pool positions (row layout [n][3])
__global__ void pool_pos(const float* __restrict__ in, float* __restrict__ out, int nOut)
{
  const int gid = blockIdx.x * blockDim.x + threadIdx.x;   // 3*nOut
  if (gid >= 3 * nOut) return;
  const int j = gid / 3, d = gid - 3 * j;
  out[gid] = 0.5f * (in[(size_t)(2 * j) * 3 + d] + in[(size_t)(2 * j + 1) * 3 + d]);
}

// u[i] = fine[i] + coarse[i/2]   (planes [3][n], coarse [3][n/2])
__global__ void upadd(const float* __restrict__ fine, const float* __restrict__ coarse,
                      float* __restrict__ out, int n)
{
  const int gid = blockIdx.x * blockDim.x + threadIdx.x;   // 3*n
  if (gid >= 3 * n) return;
  const int d = gid / n, i = gid - d * n;
  out[gid] = fine[gid] + coarse[(size_t)d * (n >> 1) + (i >> 1)];
}

__global__ void plane_from_v(const float* __restrict__ v, float* __restrict__ f, int n)
{
  const int gid = blockIdx.x * blockDim.x + threadIdx.x;   // 3*n
  if (gid >= 3 * n) return;
  const int d = gid / n, i = gid - d * n;
  f[gid] = v[(size_t)i * 3 + d];
}

__global__ void row_from_plane(const float* __restrict__ f, float* __restrict__ out, int n)
{
  const int gid = blockIdx.x * blockDim.x + threadIdx.x;   // n*3
  if (gid >= 3 * n) return;
  const int i = gid / 3, d = gid - 3 * i;
  out[gid] = f[(size_t)d * n + i];
}

// ---------------------------------------------------------------------------
// Host-side orchestration
// ---------------------------------------------------------------------------
static void edge_dispatch(int level, const float* Qb, const float* Kb,
                          const float* Vb, const float* Sb, const float* wr,
                          const float* pos, const int* src,
                          float* out, int n, int cout, hipStream_t stream)
{
  dim3 grid((n + 7) / 8);
  if (level == 0)
    edge_attn<8, 0><<<grid, 256, 0, stream>>>(Qb, Kb, Vb, Sb, wr, pos, src, out, n, cout);
  else if (level == 1)
    edge_attn<16, 1><<<grid, 256, 0, stream>>>(Qb, Kb, Vb, Sb, wr, pos, src, out, n, cout);
  else
    edge_attn<32, 2><<<grid, 256, 0, stream>>>(Qb, Kb, Vb, Sb, wr, pos, src, out, n, cout);
}

// Layer leaves per transformer (insertion order):
//  0..4  l1: Wq Wk Wv wr Ws      5  b1
//  6..10 l2: Wq Wk Wv wr Ws      11 b2
// 12..16 lo: Wq Wk Wv wr Ws
// Swizzled bf16 weight slots per transformer: [Wq2 Wk2 Wv2 Ws2 Wqo Wko] * 1024
static void run_tf(const float* const* L, const __bf16* Wsw,
                   const float* fin, const float* pos,
                   int level, int n, float* fout,
                   float* Qb, float* Kb, float* Vb, float* Sb,
                   float* hid, __bf16* hidb,
                   const int* src, hipStream_t stream)
{
  const int thr = 256;
  // layer 1 (cin=1 -> C)
  outer1<<<dim3((n * CH) / thr), thr, 0, stream>>>(fin, L[0], L[1], L[2], L[4], Qb, Kb, Vb, Sb, n);
  edge_dispatch(level, Qb, Kb, Vb, Sb, L[3], pos, src, hid, n, CH, stream);
  normbias_bf<<<dim3((n * CH) / thr), thr, 0, stream>>>(hid, L[5], hidb, n);
  // layer 2 (C -> C): four WMMA GEMMs per dim, B held in registers across dims
  gemm4_wmma<<<dim3(n / 16), 256, 0, stream>>>(hidb, Wsw + 0 * 1024, Wsw + 1 * 1024,
                                               Wsw + 2 * 1024, Wsw + 3 * 1024,
                                               Qb, Kb, Vb, Sb, n, 4);
  edge_dispatch(level, Qb, Kb, Vb, Sb, L[9], pos, src, hid, n, CH, stream);
  normbias_bf<<<dim3((n * CH) / thr), thr, 0, stream>>>(hid, L[11], hidb, n);
  // output layer (C -> 1): Q/K via WMMA, Val/S via matvec
  gemm4_wmma<<<dim3(n / 16), 256, 0, stream>>>(hidb, Wsw + 4 * 1024, Wsw + 5 * 1024,
                                               nullptr, nullptr,
                                               Qb, Kb, nullptr, nullptr, n, 2);
  matvec2<<<dim3((3 * n + thr - 1) / thr), thr, 0, stream>>>(hidb, L[14], L[16], Vb, Sb, n);
  edge_dispatch(level, Qb, Kb, Vb, Sb, L[15], pos, src, fout, n, 1, stream);
}

extern "C" void kernel_launch(void* const* d_in, const int* in_sizes, int n_in,
                              void* d_out, int out_size, void* d_ws, size_t ws_size,
                              hipStream_t stream)
{
  if (n_in < 3 + 5 * 17) return;
  const float* v   = (const float*)d_in[0];
  const float* pos = (const float*)d_in[1];
  const int*   src = (const int*)d_in[2];
  const float* P[5 * 17];
  for (int i = 0; i < 5 * 17; ++i) P[i] = (const float*)d_in[3 + i];

  // byte bump allocator over workspace (256B aligned blocks)
  char* wp = (char*)d_ws;
  auto alloc = [&](size_t bytes) { void* p = (void*)wp; wp += (bytes + 255) & ~(size_t)255; return p; };

  const int N = NN, N1 = NN / 2, N2 = NN / 4;
  float*  Qb   = (float*)alloc((size_t)3 * N * CH * 4);
  float*  Kb   = (float*)alloc((size_t)3 * N * CH * 4);
  float*  Vb   = (float*)alloc((size_t)3 * N * CH * 4);
  float*  Sb   = (float*)alloc((size_t)3 * N * CH * 4);
  float*  hid  = (float*)alloc((size_t)3 * N * CH * 4);
  __bf16* hidb = (__bf16*)alloc((size_t)3 * N * CH * 2);
  __bf16* Wsw  = (__bf16*)alloc((size_t)5 * 6 * 1024 * 2);
  float* f0   = (float*)alloc((size_t)3 * N * 4);
  float* h0   = (float*)alloc((size_t)3 * N * 4);
  float* h1f  = (float*)alloc((size_t)3 * N1 * 4);
  float* g1   = (float*)alloc((size_t)3 * N1 * 4);
  float* h2f  = (float*)alloc((size_t)3 * N2 * 4);
  float* bb   = (float*)alloc((size_t)3 * N2 * 4);
  float* u1   = (float*)alloc((size_t)3 * N1 * 4);
  float* u1t  = (float*)alloc((size_t)3 * N1 * 4);
  float* u0   = (float*)alloc((size_t)3 * N * 4);
  float* u0t  = (float*)alloc((size_t)3 * N * 4);
  float* pos1 = (float*)alloc((size_t)3 * N1 * 4);
  float* pos2 = (float*)alloc((size_t)3 * N2 * 4);
  (void)ws_size; (void)in_sizes; (void)out_size;

  const int thr = 256;

  // one-time bf16 weight pre-swizzle for all 5 transformers (30 matrices)
  for (int t = 0; t < 5; ++t) {
    const float* const* L = P + t * 17;
    swizzle_w6<<<dim3(6), 64, 0, stream>>>(L[6], L[7], L[8], L[10], L[12], L[13],
                                           Wsw + (size_t)t * 6 * 1024);
  }

  plane_from_v<<<dim3((3 * N) / thr), thr, 0, stream>>>(v, f0, N);

  // ---- down level 0 ----
  run_tf(P + 0 * 17, Wsw + 0 * 6144, f0, pos, 0, N, h0, Qb, Kb, Vb, Sb, hid, hidb, src, stream);
  pool_feat<<<dim3((3 * N1 + thr - 1) / thr), thr, 0, stream>>>(h0, h1f, N1);
  pool_pos <<<dim3((3 * N1 + thr - 1) / thr), thr, 0, stream>>>(pos, pos1, N1);

  // ---- down level 1 ----
  run_tf(P + 1 * 17, Wsw + 1 * 6144, h1f, pos1, 1, N1, g1, Qb, Kb, Vb, Sb, hid, hidb, src, stream);
  pool_feat<<<dim3((3 * N2 + thr - 1) / thr), thr, 0, stream>>>(g1, h2f, N2);
  pool_pos <<<dim3((3 * N2 + thr - 1) / thr), thr, 0, stream>>>(pos1, pos2, N2);

  // ---- bottleneck ----
  run_tf(P + 2 * 17, Wsw + 2 * 6144, h2f, pos2, 2, N2, bb, Qb, Kb, Vb, Sb, hid, hidb, src, stream);

  // ---- up level 1 ----
  upadd<<<dim3((3 * N1 + thr - 1) / thr), thr, 0, stream>>>(g1, bb, u1, N1);
  run_tf(P + 3 * 17, Wsw + 3 * 6144, u1, pos1, 1, N1, u1t, Qb, Kb, Vb, Sb, hid, hidb, src, stream);

  // ---- up level 0 ----
  upadd<<<dim3((3 * N + thr - 1) / thr), thr, 0, stream>>>(h0, u1t, u0, N);
  run_tf(P + 4 * 17, Wsw + 4 * 6144, u0, pos, 0, N, u0t, Qb, Kb, Vb, Sb, hid, hidb, src, stream);

  row_from_plane<<<dim3((3 * N) / thr), thr, 0, stream>>>(u0t, (float*)d_out, N);
}
